// AttentionSessionGNN_40793599377664
// MI455X (gfx1250) — matrix operand order repaired and checked by reading, not compiled, not run on hardware
//
#include <hip/hip_runtime.h>
#include <math.h>

typedef __attribute__((ext_vector_type(16))) __bf16 v16bf;
typedef __attribute__((ext_vector_type(8)))  __bf16 v8bf;
typedef __attribute__((ext_vector_type(8)))  float  v8f;

#define BGRAPHS 4096
#define NN 50
#define EE 100
#define DD 128

// ---------------- f32 -> bf16 bulk convert (8 elems/thread, b128-friendly) ----------------
__global__ void k_cvt_bf16(const float* __restrict__ src, __bf16* __restrict__ dst, int n8) {
  int t = blockIdx.x * blockDim.x + threadIdx.x;
  if (t >= n8) return;
  const float* s = src + (size_t)t * 8;
  v8bf o;
#pragma unroll
  for (int i = 0; i < 8; ++i) o[i] = (__bf16)s[i];
  *(v8bf*)(dst + (size_t)t * 8) = o;
}

// ---------------- embedding gather: x = item_emb[x_idx] (f32 + bf16 copies) --------------
__global__ void k_embed(const int* __restrict__ x_idx, const float* __restrict__ emb,
                        float* __restrict__ x, __bf16* __restrict__ xb, int total) {
  int t = blockIdx.x * blockDim.x + threadIdx.x;
  if (t >= total) return;
  int row = t >> 7;
  int d   = t & 127;
  float v = emb[(size_t)x_idx[row] * DD + d];
  x[t]  = v;
  xb[t] = (__bf16)v;
}

// ---------------- edge working-copy init ----------------
__global__ void k_edge_init(const int* __restrict__ edge_index, int* __restrict__ src,
                            int* __restrict__ dst, float* __restrict__ emask, int ecnt) {
  int t = blockIdx.x * blockDim.x + threadIdx.x;
  if (t >= ecnt) return;
  src[t]   = edge_index[t];
  dst[t]   = edge_index[ecnt + t];
  emask[t] = 1.0f;
}

// ---------------- per-graph masked segment mean -> bf16 (block = graph, 128 thr) --------
__global__ __launch_bounds__(128) void k_seg_mean(
    const float* __restrict__ x, const int* __restrict__ src,
    const int* __restrict__ dst, const float* __restrict__ emask,
    __bf16* __restrict__ meanb, int n_per) {
  __shared__ float agg[NN * DD];
  __shared__ float cnt[NN];
  int b = blockIdx.x;
  int tid = threadIdx.x;                 // 0..127 == feature lane
  for (int i = tid; i < n_per * DD; i += DD) agg[i] = 0.0f;
  if (tid < n_per) cnt[tid] = 0.0f;
  __syncthreads();
  int ebase = b * EE;
  int nbase = b * n_per;
  for (int e = 0; e < EE; ++e) {
    float em = emask[ebase + e];
    if (em > 0.0f) {
      int sg = src[ebase + e];
      int dl = dst[ebase + e] - nbase;   // edges never cross graphs
      agg[dl * DD + tid] += x[(size_t)sg * DD + tid] * em;  // lane-owned -> no race
      if (tid == 0) cnt[dl] += em;
    }
  }
  __syncthreads();
  for (int n = 0; n < n_per; ++n)
    meanb[(size_t)(nbase + n) * DD + tid] =
        (__bf16)(agg[n * DD + tid] / fmaxf(cnt[n], 1.0f));
}

// ---------------- bf16 WMMA fragment loaders (pure b128 loads) ----------------
__device__ __forceinline__ v16bf load_a_frag(const __bf16* __restrict__ A, int m, int K,
                                             int kb, int a_koff) {
  const __bf16* p = A + (size_t)m * K + kb + a_koff;
  v8bf lo = *(const v8bf*)p;            // K = kb+a_koff+0..7
  v8bf hi = *(const v8bf*)(p + 16);     // K = kb+16+a_koff+0..7
  return __builtin_shufflevector(lo, hi, 0, 1, 2, 3, 4, 5, 6, 7,
                                 8, 9, 10, 11, 12, 13, 14, 15);
}
__device__ __forceinline__ v16bf load_b_frag(const __bf16* __restrict__ W, int n, int K,
                                             int kb, int b_koff) {
  const __bf16* p = W + (size_t)n * K + kb + b_koff;
  v8bf lo = *(const v8bf*)p;            // K = kb+b_koff+0..7
  v8bf hi = *(const v8bf*)(p + 16);     // K = kb+16+b_koff+0..7
  return __builtin_shufflevector(lo, hi, 0, 1, 2, 3, 4, 5, 6, 7,
                                 8, 9, 10, 11, 12, 13, 14, 15);
}

// ---------------- WMMA GEMM: C[m,n] = sum_k A[m,k]*W[n,k] (+A2*W2) + bias[n] ------------
// Wave owns a 16x128 strip (8 N-tiles); K is compile-time -> fully unrolled, loads
// scheduled ahead of the consuming v_wmma. M mult of 16, N mult of 128, K in {128,256}.
template <bool DUAL, int K>
__global__ __launch_bounds__(256) void k_gemm_bf16(
    const __bf16* __restrict__ A,  const __bf16* __restrict__ W,
    const __bf16* __restrict__ A2, const __bf16* __restrict__ W2,
    const float* __restrict__ bias, float* __restrict__ C,
    int M, int N, int relu) {
  int wave = (blockIdx.x * blockDim.x + threadIdx.x) >> 5;
  int lane = threadIdx.x & 31;
  int mtiles  = M >> 4;
  int ngroups = N >> 7;                  // 8 tiles of 16 per group
  if (wave >= mtiles * ngroups) return;  // wave-uniform; EXEC all-ones for WMMA
  int mt = wave % mtiles;
  int ng = wave / mtiles;
  int m  = (mt << 4) + (lane & 15);
  int n0 = (ng << 7) + (lane & 15);
  int a_koff = (lane & 16) ? 8  : 0;     // 16-bit A 16x32 VGPR layout
  int b_koff = (lane & 16) ? 8  : 0;     // 16-bit B 32x16 VGPR layout
  v8f acc[8];
#pragma unroll
  for (int t = 0; t < 8; ++t) acc[t] = (v8f){0.f,0.f,0.f,0.f,0.f,0.f,0.f,0.f};
#pragma unroll
  for (int kb = 0; kb < K; kb += 32) {
    v16bf af = load_a_frag(A, m, K, kb, a_koff);
#pragma unroll
    for (int t = 0; t < 8; ++t)
      acc[t] = __builtin_amdgcn_wmma_f32_16x16x32_bf16(false, af, false,
                 load_b_frag(W, n0 + (t << 4), K, kb, b_koff), (short)0, acc[t],
                 false, false);
    if (DUAL) {
      v16bf af2 = load_a_frag(A2, m, K, kb, a_koff);
#pragma unroll
      for (int t = 0; t < 8; ++t)
        acc[t] = __builtin_amdgcn_wmma_f32_16x16x32_bf16(false, af2, false,
                   load_b_frag(W2, n0 + (t << 4), K, kb, b_koff), (short)0, acc[t],
                   false, false);
    }
  }
  int mrow = (mt << 4) + ((lane & 16) ? 8 : 0);   // f32 C/D layout
#pragma unroll
  for (int t = 0; t < 8; ++t) {
    int n = n0 + (t << 4);
    float bn = bias ? bias[n] : 0.0f;
#pragma unroll
    for (int v = 0; v < 8; ++v) {
      float val = acc[t][v] + bn;
      if (relu) val = fmaxf(val, 0.0f);
      C[(size_t)(mrow + v) * N + n] = val;
    }
  }
}

// ---------------- row-normalize + relu (in place) + pooling score ----------------
__global__ __launch_bounds__(128) void k_norm_score(
    float* __restrict__ h, const float* __restrict__ w, float* __restrict__ score) {
  __shared__ float red[DD];
  int m = blockIdx.x;
  int tid = threadIdx.x;
  float v = h[(size_t)m * DD + tid];
  red[tid] = v * v;
  __syncthreads();
  for (int s = 64; s > 0; s >>= 1) { if (tid < s) red[tid] += red[tid + s]; __syncthreads(); }
  float norm = sqrtf(red[0]);
  __syncthreads();
  float xv = fmaxf(v / fmaxf(norm, 1e-12f), 0.0f);
  h[(size_t)m * DD + tid] = xv;
  float wd = w[tid];
  red[tid] = wd * wd;
  __syncthreads();
  for (int s = 64; s > 0; s >>= 1) { if (tid < s) red[tid] += red[tid + s]; __syncthreads(); }
  float wn = sqrtf(red[0]);
  __syncthreads();
  red[tid] = xv * wd;
  __syncthreads();
  for (int s = 64; s > 0; s >>= 1) { if (tid < s) red[tid] += red[tid + s]; __syncthreads(); }
  if (tid == 0) score[m] = tanhf(red[0] / wn);
}

__global__ void k_newid_init(int* __restrict__ new_id, int total) {
  int t = blockIdx.x * blockDim.x + threadIdx.x;
  if (t < total) new_id[t] = -1;
}

// ---------------- per-graph top-k (matches lax.top_k tie-breaking) ----------------
__global__ void k_topk(const float* __restrict__ score, float* __restrict__ vals,
                       int* __restrict__ perm, int* __restrict__ new_id,
                       int n_per, int kk) {
  int b = blockIdx.x * blockDim.x + threadIdx.x;
  if (b >= BGRAPHS) return;
  float s[NN];
  int used[NN];
  for (int i = 0; i < n_per; ++i) { s[i] = score[b * n_per + i]; used[i] = 0; }
  for (int j = 0; j < kk; ++j) {
    int best = 0; float bv = -1e30f;
    for (int i = 0; i < n_per; ++i)
      if (!used[i] && s[i] > bv) { bv = s[i]; best = i; }
    used[best] = 1;
    vals[b * kk + j] = bv;
    int g = b * n_per + best;
    perm[b * kk + j] = g;
    new_id[g] = b * kk + j;
  }
}

// ---------------- gather kept nodes, gate by score (f32 + bf16 copies) ----------------
__global__ void k_gather(const float* __restrict__ xin, const float* __restrict__ vals,
                         const int* __restrict__ perm, float* __restrict__ xout,
                         __bf16* __restrict__ xbout, int rows) {
  int t = blockIdx.x * blockDim.x + threadIdx.x;
  if (t >= rows * DD) return;
  int r = t >> 7, d = t & 127;
  float v = xin[(size_t)perm[r] * DD + d] * vals[r];
  xout[t]  = v;
  xbout[t] = (__bf16)v;
}

// ---------------- edge remap / mask ----------------
__global__ void k_remap(int* __restrict__ src, int* __restrict__ dst,
                        float* __restrict__ emask, const int* __restrict__ new_id, int ecnt) {
  int t = blockIdx.x * blockDim.x + threadIdx.x;
  if (t >= ecnt) return;
  int ns = new_id[src[t]];
  int nd = new_id[dst[t]];
  bool valid = (ns >= 0) && (nd >= 0);
  emask[t] = valid ? emask[t] : 0.0f;
  src[t] = valid ? ns : 0;
  dst[t] = valid ? nd : 0;
}

// ---------------- global mean/max pool -> per-layer features ----------------
__global__ __launch_bounds__(128) void k_pool(const float* __restrict__ x,
                                              float* __restrict__ feats, int kk, int layer) {
  int b = blockIdx.x;
  int d = threadIdx.x;
  float sum = 0.0f, mx = -1e30f;
  for (int j = 0; j < kk; ++j) {
    float v = x[(size_t)(b * kk + j) * DD + d];
    sum += v;
    mx = fmaxf(mx, v);
  }
  float* fb = feats + (size_t)(b * 3 + layer) * 256;
  fb[d] = sum / (float)kk;
  fb[DD + d] = mx;
}

// ---------------- 3-token, 4-head attention core (thread = (graph, head)) ----------------
__global__ void k_attn(const float* __restrict__ qkv, float* __restrict__ o) {
  int t = blockIdx.x * blockDim.x + threadIdx.x;
  if (t >= BGRAPHS * 4) return;
  int b = t >> 2, h = t & 3;
  const float* base = qkv + (size_t)b * 3 * 768;
  float sc[3][3];
  for (int i = 0; i < 3; ++i)
    for (int j = 0; j < 3; ++j) {
      float acc = 0.0f;
      for (int d = 0; d < 64; ++d)
        acc += base[i * 768 + h * 64 + d] * base[j * 768 + 256 + h * 64 + d];
      sc[i][j] = acc * 0.125f;            // 1/sqrt(64)
    }
  for (int i = 0; i < 3; ++i) {
    float mx = fmaxf(sc[i][0], fmaxf(sc[i][1], sc[i][2]));
    float e0 = expf(sc[i][0] - mx), e1 = expf(sc[i][1] - mx), e2 = expf(sc[i][2] - mx);
    float inv = 1.0f / (e0 + e1 + e2);
    sc[i][0] = e0 * inv; sc[i][1] = e1 * inv; sc[i][2] = e2 * inv;
  }
  for (int i = 0; i < 3; ++i)
    for (int d = 0; d < 64; ++d) {
      float acc = sc[i][0] * base[0 * 768 + 512 + h * 64 + d]
                + sc[i][1] * base[1 * 768 + 512 + h * 64 + d]
                + sc[i][2] * base[2 * 768 + 512 + h * 64 + d];
      o[(size_t)(b * 3 + i) * 256 + h * 64 + d] = acc;
    }
}

// ---------------- mean over 3 tokens ----------------
__global__ void k_token_mean(const float* __restrict__ o2, float* __restrict__ pooled, int total) {
  int t = blockIdx.x * blockDim.x + threadIdx.x;
  if (t >= total) return;
  int b = t >> 8, d = t & 255;
  pooled[t] = (o2[(size_t)(b * 3 + 0) * 256 + d] +
               o2[(size_t)(b * 3 + 1) * 256 + d] +
               o2[(size_t)(b * 3 + 2) * 256 + d]) * (1.0f / 3.0f);
}

// ---------------- final logit + sigmoid ----------------
__global__ void k_final(const float* __restrict__ h1, const float* __restrict__ W2,
                        const float* __restrict__ b2, float* __restrict__ out) {
  int b = blockIdx.x * blockDim.x + threadIdx.x;
  if (b >= BGRAPHS) return;
  float acc = b2[0];
  for (int d = 0; d < DD; ++d) acc += h1[(size_t)b * DD + d] * W2[d];
  out[b] = 1.0f / (1.0f + expf(-acc));
}

extern "C" void kernel_launch(void* const* d_in, const int* in_sizes, int n_in,
                              void* d_out, int out_size, void* d_ws, size_t ws_size,
                              hipStream_t stream) {
  const int*   x_idx      = (const int*)d_in[0];
  const int*   edge_index = (const int*)d_in[1];
  const float* item_emb   = (const float*)d_in[2];
  const float* sage_Wl    = (const float*)d_in[3];
  const float* sage_bl    = (const float*)d_in[4];
  const float* sage_Wr    = (const float*)d_in[5];
  const float* pool_w     = (const float*)d_in[6];
  const float* attn_Win   = (const float*)d_in[7];
  const float* attn_bin   = (const float*)d_in[8];
  const float* attn_Wout  = (const float*)d_in[9];
  const float* attn_bout  = (const float*)d_in[10];
  const float* cls_W1     = (const float*)d_in[11];
  const float* cls_b1     = (const float*)d_in[12];
  const float* cls_W2     = (const float*)d_in[13];
  const float* cls_b2     = (const float*)d_in[14];
  float* out = (float*)d_out;

  char* p = (char*)d_ws;
  auto carve = [&](size_t bytes) { char* r = p; p += (bytes + 255) & ~(size_t)255; return r; };
  const size_t S = (size_t)BGRAPHS * NN * DD * sizeof(float);     // 104.86 MB
  float*  P0    = (float*)carve(S);              // x f32 (tail: attn o, pooled)
  float*  P2    = (float*)carve(S);              // h f32 (tail: qkv, o2, h1)
  __bf16* XB    = (__bf16*)carve(S / 2);         // x bf16 (tail: pooled bf16)
  __bf16* MB    = (__bf16*)carve(S / 2);         // mean bf16 (tail: o bf16)
  float*  score = (float*)carve((size_t)BGRAPHS * NN * 4);
  float*  vals  = (float*)carve((size_t)BGRAPHS * NN * 4);
  int*    perm  = (int*)carve((size_t)BGRAPHS * NN * 4);
  int*    newid = (int*)carve((size_t)BGRAPHS * NN * 4);
  int*    srcW  = (int*)carve((size_t)BGRAPHS * EE * 4);
  int*    dstW  = (int*)carve((size_t)BGRAPHS * EE * 4);
  float*  emask = (float*)carve((size_t)BGRAPHS * EE * 4);
  float*  feats = (float*)carve((size_t)BGRAPHS * 3 * 256 * 4);
  __bf16* FB    = (__bf16*)carve((size_t)BGRAPHS * 3 * 256 * 2);
  __bf16* wlb   = (__bf16*)carve((size_t)3 * DD * DD * 2);
  __bf16* wrb   = (__bf16*)carve((size_t)3 * DD * DD * 2);
  __bf16* winb  = (__bf16*)carve((size_t)768 * 256 * 2);
  __bf16* woutb = (__bf16*)carve((size_t)256 * 256 * 2);
  __bf16* w1b   = (__bf16*)carve((size_t)DD * 256 * 2);

  // --- convert all weights to bf16 once ---
  k_cvt_bf16<<<(3 * DD * DD / 8 + 255) / 256, 256, 0, stream>>>(sage_Wl, wlb, 3 * DD * DD / 8);
  k_cvt_bf16<<<(3 * DD * DD / 8 + 255) / 256, 256, 0, stream>>>(sage_Wr, wrb, 3 * DD * DD / 8);
  k_cvt_bf16<<<(768 * 256 / 8 + 255) / 256, 256, 0, stream>>>(attn_Win, winb, 768 * 256 / 8);
  k_cvt_bf16<<<(256 * 256 / 8 + 255) / 256, 256, 0, stream>>>(attn_Wout, woutb, 256 * 256 / 8);
  k_cvt_bf16<<<(DD * 256 / 8 + 255) / 256, 256, 0, stream>>>(cls_W1, w1b, DD * 256 / 8);

  const int ecnt = BGRAPHS * EE;
  const int totalX = BGRAPHS * NN * DD;
  k_embed<<<(totalX + 255) / 256, 256, 0, stream>>>(x_idx, item_emb, P0, XB, totalX);
  k_edge_init<<<(ecnt + 255) / 256, 256, 0, stream>>>(edge_index, srcW, dstW, emask, ecnt);

  int n_per = NN;
  for (int layer = 0; layer < 3; ++layer) {
    int n_nodes = BGRAPHS * n_per;
    k_seg_mean<<<BGRAPHS, 128, 0, stream>>>(P0, srcW, dstW, emask, MB, n_per);
    int waves = (n_nodes / 16) * (DD / 128);
    k_gemm_bf16<true, DD><<<(waves * 32 + 255) / 256, 256, 0, stream>>>(
        MB, wlb + (size_t)layer * DD * DD,
        XB, wrb + (size_t)layer * DD * DD,
        sage_bl + layer * DD, P2, n_nodes, DD, 0);
    k_norm_score<<<n_nodes, 128, 0, stream>>>(P2, pool_w + layer * DD, score);
    int kk = (n_per * 4 + 4) / 5;   // ceil(0.8 * n_per): 40, 32, 26
    k_newid_init<<<(n_nodes + 255) / 256, 256, 0, stream>>>(newid, n_nodes);
    k_topk<<<(BGRAPHS + 63) / 64, 64, 0, stream>>>(score, vals, perm, newid, n_per, kk);
    int nrows = BGRAPHS * kk;
    k_gather<<<(nrows * DD + 255) / 256, 256, 0, stream>>>(P2, vals, perm, P0, XB, nrows);
    k_remap<<<(ecnt + 255) / 256, 256, 0, stream>>>(srcW, dstW, emask, newid, ecnt);
    k_pool<<<BGRAPHS, 128, 0, stream>>>(P0, feats, kk, layer);
    n_per = kk;
  }

  // ---- attention tail ----
  const int Mq = BGRAPHS * 3;                                          // 12288
  k_cvt_bf16<<<(Mq * 256 / 8 + 255) / 256, 256, 0, stream>>>(feats, FB, Mq * 256 / 8);
  int waves = (Mq / 16) * (768 / 128);
  k_gemm_bf16<false, 256><<<(waves * 32 + 255) / 256, 256, 0, stream>>>(
      FB, winb, nullptr, nullptr, attn_bin, P2, Mq, 768, 0);           // qkv -> P2
  k_attn<<<(BGRAPHS * 4 + 255) / 256, 256, 0, stream>>>(P2, P0);       // o -> P0
  k_cvt_bf16<<<(Mq * 256 / 8 + 255) / 256, 256, 0, stream>>>(P0, MB, Mq * 256 / 8);
  waves = (Mq / 16) * (256 / 128);
  k_gemm_bf16<false, 256><<<(waves * 32 + 255) / 256, 256, 0, stream>>>(
      MB, woutb, nullptr, nullptr, attn_bout, P2, Mq, 256, 0);         // o2 -> P2
  k_token_mean<<<(BGRAPHS * 256 + 255) / 256, 256, 0, stream>>>(P2, P0, BGRAPHS * 256);
  k_cvt_bf16<<<(BGRAPHS * 256 / 8 + 255) / 256, 256, 0, stream>>>(P0, XB, BGRAPHS * 256 / 8);
  waves = (BGRAPHS / 16) * (DD / 128);
  k_gemm_bf16<false, 256><<<(waves * 32 + 255) / 256, 256, 0, stream>>>(
      XB, w1b, nullptr, nullptr, cls_b1, P2, BGRAPHS, DD, 1);          // h1 -> P2
  k_final<<<(BGRAPHS + 255) / 256, 256, 0, stream>>>(P2, cls_W2, cls_b2, out);
}
